// GroupQueryAttention_71708773974527
// MI455X (gfx1250) — matrix operand order repaired and checked by reading, not compiled, not run on hardware
//
#include <hip/hip_runtime.h>
#include <hip/hip_bf16.h>

// ---------------------------------------------------------------------------
// GQA block for MI455X (gfx1250, wave32, WMMA + Tensor Data Mover).
// B=2, S=2048, D_IN=D_OUT=2048, 32 Q heads, 8 KV heads, head_dim=64, GROUP=4.
// bf16 WMMA (16x16x32) with fp32 accumulation; TDM double-buffered GEMM tiles;
// fp32 RoPE/softmax. Compute-bound -> matrix cores drive the roofline.
// ---------------------------------------------------------------------------

typedef __attribute__((ext_vector_type(16))) __bf16 v16bf;
typedef __attribute__((ext_vector_type(8)))  float  v8f;
typedef unsigned int u32x4 __attribute__((ext_vector_type(4)));
typedef int          i32x8 __attribute__((ext_vector_type(8)));
typedef int          i32x4 __attribute__((ext_vector_type(4)));

union FragB16 { v16bf v; unsigned u[8]; };
union FragF32 { v8f  v; float    f[8]; };

static constexpr int BB   = 2;
static constexpr int SS   = 2048;
static constexpr int DIN  = 2048;
static constexpr int DOUT = 2048;
static constexpr int NH   = 32;
static constexpr int NKV  = 8;
static constexpr int HD   = 64;

__device__ __forceinline__ float bf2f(__bf16 x) { return (float)x; }

// A-matrix (16x32 bf16) per-lane K index for VGPR j (pair k,k+1).
__device__ __forceinline__ int a_kidx(int j, int lh) {
    return ((j < 4) ? 2 * j : 16 + 2 * (j - 4)) + lh * 8;
}
// B-matrix (32x16 bf16) per-lane K index for VGPR j (pair k,k+1).
__device__ __forceinline__ int b_kidx(int j, int lh) {
    return ((j < 4) ? 2 * j : 16 + 2 * (j - 4)) + lh * 8;
}

// LDS byte offset of a __shared__ object: flat-address low 32 bits are the
// LDS logical address (ISA 10.2).
__device__ __forceinline__ unsigned lds_off(const void* p) {
    return (unsigned)(size_t)p;
}

// ---------------------------------------------------------------------------
// TDM 2D tile load: global (row-major, elem stride0 per row) -> LDS,
// tile_d0 contiguous bf16 elements per row, tile_d1 rows. D# per ISA ch.8.
// Issued once per calling wave; tracked by TENSORcnt.
// (6-arg toolchain form: groups 0..3 + extra group + cpol.)
// ---------------------------------------------------------------------------
__device__ __forceinline__ void tdm_load_2d_bf16(const __bf16* gptr, unsigned ldsByteOff,
                                                 int tile_d0, int tile_d1, int stride0) {
    unsigned long long ga = (unsigned long long)(size_t)gptr;
    const unsigned td0 = (unsigned)stride0;      // tensor_dim0 (elems)
    const unsigned td1 = 0x00100000u;            // tensor_dim1: generous (no OOB)
    u32x4 g0;
    g0[0] = 1u;                                  // count=1 valid, user mode
    g0[1] = ldsByteOff;                          // lds_addr
    g0[2] = (unsigned)(ga & 0xFFFFFFFFu);        // global_addr[31:0]
    g0[3] = (unsigned)((ga >> 32) & 0x01FFFFFFu) | (2u << 30);  // addr[56:32], type=2
    i32x8 g1;
    g1[0] = (int)(1u << 16);                                       // data_size=1 (2B)
    g1[1] = (int)((td0 & 0xFFFFu) << 16);                          // dim0[15:0]
    g1[2] = (int)(((td0 >> 16) & 0xFFFFu) | ((td1 & 0xFFFFu) << 16));   // dim0[31:16], dim1[15:0]
    g1[3] = (int)(((td1 >> 16) & 0xFFFFu) | ((unsigned)tile_d0 << 16)); // dim1[31:16], tile_dim0
    g1[4] = (int)((unsigned)tile_d1 & 0xFFFFu);                    // tile_dim1 (tile_dim2=0)
    g1[5] = stride0;                                               // dim0_stride[31:0]
    g1[6] = 0;                                                     // dim0_stride[47:32]
    g1[7] = 0;
    i32x4 z4 = {0, 0, 0, 0};
    i32x8 z8 = {0, 0, 0, 0, 0, 0, 0, 0};
    __builtin_amdgcn_tensor_load_to_lds(g0, g1, z4, z4, z8, 0);
}

// ---------------------------------------------------------------------------
// fp32 -> bf16 elementwise convert (vectorized x4)
// ---------------------------------------------------------------------------
__global__ __launch_bounds__(256) void f32_to_bf16_kernel(const float* __restrict__ in,
                                                          __bf16* __restrict__ out,
                                                          long n4) {
    long i = (long)blockIdx.x * blockDim.x + threadIdx.x;
    if (i >= n4) return;
    float4 v = ((const float4*)in)[i];
    __bf16 o[4] = {(__bf16)v.x, (__bf16)v.y, (__bf16)v.z, (__bf16)v.w};
    *(uint2*)&out[i * 4] = *(const uint2*)o;
}

// ---------------------------------------------------------------------------
// fp32 [K][N] -> bf16 transposed [N][K] (32x32 LDS tile, coalesced both sides)
// ---------------------------------------------------------------------------
__global__ __launch_bounds__(256) void f32_to_bf16_transpose_kernel(const float* __restrict__ in,
                                                                    __bf16* __restrict__ out,
                                                                    int K, int N) {
    __shared__ float tile[32][33];
    const int k0 = blockIdx.y * 32, n0 = blockIdx.x * 32;
    const int tx = threadIdx.x & 31, ty = threadIdx.x >> 5;  // 32 x 8
#pragma unroll
    for (int i = ty; i < 32; i += 8)
        tile[i][tx] = in[(size_t)(k0 + i) * N + n0 + tx];
    __syncthreads();
#pragma unroll
    for (int i = ty; i < 32; i += 8)
        out[(size_t)(n0 + i) * K + k0 + tx] = (__bf16)tile[tx][i];
}

// ---------------------------------------------------------------------------
// bf16 GEMM: C[M,N] = A[M,K] @ Bt[N,K]^T (Bt stored N-major). Block tile
// 128x64, 8 waves (4x2), wave tile 32x32 (2x2 WMMA). K-step 64, TDM
// double-buffered LDS staging overlapped with WMMA via TENSORcnt.
// ---------------------------------------------------------------------------
template <bool STORE_F32>
__global__ __launch_bounds__(256) void gemm_bf16_wmma(const __bf16* __restrict__ A,
                                                      const __bf16* __restrict__ Bt,
                                                      void* __restrict__ Cout,
                                                      int M, int N, int K) {
    constexpr int KC = 64;
    __shared__ __align__(16) __bf16 sA[2][128 * KC];  // [row][k]
    __shared__ __align__(16) __bf16 sB[2][64 * KC];   // [col][k]

    const int tid  = threadIdx.x;
    const int lane = tid & 31, wave = tid >> 5;
    const int lh   = lane >> 4, l15 = lane & 15;
    const int wm   = wave >> 1, wn = wave & 1;        // 4 x 2 wave grid
    const int bm   = blockIdx.y * 128, bn = blockIdx.x * 64;

    FragF32 acc[2][2];
#pragma unroll
    for (int mt = 0; mt < 2; ++mt)
#pragma unroll
        for (int nt = 0; nt < 2; ++nt)
#pragma unroll
            for (int r = 0; r < 8; ++r) acc[mt][nt].f[r] = 0.0f;

    // Prime buffer 0 (wave 0 drives the TDM; EXEC-independent DMA).
    if (wave == 0) {
        tdm_load_2d_bf16(A  + (size_t)bm * K, lds_off(&sA[0][0]), KC, 128, K);
        tdm_load_2d_bf16(Bt + (size_t)bn * K, lds_off(&sB[0][0]), KC, 64,  K);
    }

    int buf = 0;
    for (int k0 = 0; k0 < K; k0 += KC) {
        const bool more = (k0 + KC) < K;
        if (more && wave == 0) {   // prefetch next K slice into the other buffer
            tdm_load_2d_bf16(A  + (size_t)bm * K + k0 + KC, lds_off(&sA[buf ^ 1][0]), KC, 128, K);
            tdm_load_2d_bf16(Bt + (size_t)bn * K + k0 + KC, lds_off(&sB[buf ^ 1][0]), KC, 64,  K);
        }
        if (wave == 0) {
            if (more) __builtin_amdgcn_s_wait_tensorcnt(2);  // current pair done
            else      __builtin_amdgcn_s_wait_tensorcnt(0);
        }
        __syncthreads();

#pragma unroll
        for (int kk = 0; kk < KC; kk += 32) {
            FragB16 af[2], bf[2];
#pragma unroll
            for (int mt = 0; mt < 2; ++mt) {
                const int row = wm * 32 + mt * 16 + l15;
#pragma unroll
                for (int j = 0; j < 8; ++j)
                    af[mt].u[j] = *(const unsigned*)&sA[buf][row * KC + kk + a_kidx(j, lh)];
            }
#pragma unroll
            for (int nt = 0; nt < 2; ++nt) {
                const int col = wn * 32 + nt * 16 + l15;
#pragma unroll
                for (int j = 0; j < 8; ++j)
                    bf[nt].u[j] = *(const unsigned*)&sB[buf][col * KC + kk + b_kidx(j, lh)];
            }
#pragma unroll
            for (int mt = 0; mt < 2; ++mt)
#pragma unroll
                for (int nt = 0; nt < 2; ++nt)
                    acc[mt][nt].v = __builtin_amdgcn_wmma_f32_16x16x32_bf16(
                        false, af[mt].v, false, bf[nt].v, (short)0, acc[mt][nt].v,
                        false, false);
        }
        __syncthreads();   // all waves done reading buf before it is re-filled
        buf ^= 1;
    }

#pragma unroll
    for (int mt = 0; mt < 2; ++mt)
#pragma unroll
        for (int nt = 0; nt < 2; ++nt)
#pragma unroll
            for (int r = 0; r < 8; ++r) {
                const int row = bm + wm * 32 + mt * 16 + r + 8 * lh;
                const int col = bn + wn * 32 + nt * 16 + l15;
                if (STORE_F32)
                    ((float*)Cout)[(size_t)row * N + col] = acc[mt][nt].f[r];
                else
                    ((__bf16*)Cout)[(size_t)row * N + col] = (__bf16)acc[mt][nt].f[r];
            }
}

// ---------------------------------------------------------------------------
// RoPE + reshape: in[b*S+s][h*64+d] -> out[b][h][s][d], fp32 math.
// ---------------------------------------------------------------------------
__global__ __launch_bounds__(256) void rope_reshape_kernel(const __bf16* __restrict__ in,
                                                           const float* __restrict__ cosT,
                                                           const float* __restrict__ sinT,
                                                           __bf16* __restrict__ out,
                                                           int heads) {
    long idx = (long)blockIdx.x * blockDim.x + threadIdx.x;  // b,s,h,d (d fastest)
    int d = idx & 63;
    long t = idx >> 6;
    int h = (int)(t % heads); t /= heads;
    int s = (int)(t % SS);
    int b = (int)(t / SS);

    size_t inoff = ((size_t)(b * SS + s) * heads + h) * HD;
    float q = bf2f(in[inoff + d]);
    float p = bf2f(in[inoff + ((d < 32) ? d + 32 : d - 32)]);
    float rot = (d < 32) ? -p : p;
    float v = q * cosT[s * HD + d] + rot * sinT[s * HD + d];
    out[(((size_t)b * heads + h) * SS + s) * HD + d] = (__bf16)v;
}

// ---------------------------------------------------------------------------
// V reshape+transpose: in[b*S+s][kv*64+d] -> out[b][kv][d][s]
// ---------------------------------------------------------------------------
__global__ __launch_bounds__(256) void v_transpose_kernel(const __bf16* __restrict__ in,
                                                          __bf16* __restrict__ out) {
    long idx = (long)blockIdx.x * blockDim.x + threadIdx.x;  // b,kv,d,s (s fastest)
    int s = idx & (SS - 1);
    long t = idx >> 11;
    int d = (int)(t & 63); t >>= 6;
    int kv = (int)(t & 7);
    int b = (int)(t >> 3);
    out[idx] = in[(size_t)(b * SS + s) * (NKV * HD) + kv * HD + d];
}

// ---------------------------------------------------------------------------
// Flash-style causal GQA attention. Block = 128 thr (4 waves), one
// (b, h, 64-row q tile) per block; each wave owns a 16-row q strip (identical
// kv trip count -> no divergence; per-wave LDS -> no cross-wave barriers).
// ---------------------------------------------------------------------------
__global__ __launch_bounds__(128) void attn_kernel(const __bf16* __restrict__ Qr,
                                                   const __bf16* __restrict__ Kr,
                                                   const __bf16* __restrict__ Vt,
                                                   __bf16* __restrict__ Ctx) {
    __shared__ __align__(16) __bf16 sP[4][16 * 64];  // per-wave P tile

    const int lane = threadIdx.x & 31, wave = threadIdx.x >> 5;
    const int lh = lane >> 4, l15 = lane & 15;
    const int qt = blockIdx.x, h = blockIdx.y, b = blockIdx.z;
    const int kv = h >> 2;  // GROUP = 4
    const int q0 = qt * 64;
    const int qrow = q0 + wave * 16 + l15;

    const __bf16* Qh = Qr + (((size_t)b * NH + h) * SS) * HD;   // [s][d]
    const __bf16* Kh = Kr + (((size_t)b * NKV + kv) * SS) * HD; // [s][d]
    const __bf16* Vh = Vt + (((size_t)b * NKV + kv) * HD) * SS; // [d][s]

    // Q A-fragments (head_dim 64 -> two 16x16x32 k-steps), loaded once.
    FragB16 aq[2];
#pragma unroll
    for (int kk = 0; kk < 2; ++kk)
#pragma unroll
        for (int j = 0; j < 8; ++j) {
            const int d = kk * 32 + a_kidx(j, lh);
            aq[kk].u[j] = *(const unsigned*)&Qh[(size_t)qrow * HD + d];
        }

    float m[8], l[8];
    FragF32 o[4];
#pragma unroll
    for (int r = 0; r < 8; ++r) { m[r] = -3.0e38f; l[r] = 0.0f; }
#pragma unroll
    for (int nt = 0; nt < 4; ++nt)
#pragma unroll
        for (int r = 0; r < 8; ++r) o[nt].f[r] = 0.0f;

    const int nChunks = qt + 1;  // causal: kv chunks 0..qt
    for (int c = 0; c < nChunks; ++c) {
        const int kvb = c * 64;

        // ---- scores S = Q @ K^T over this 64-wide kv chunk ----
        FragF32 s[4];
#pragma unroll
        for (int nt = 0; nt < 4; ++nt) {
#pragma unroll
            for (int r = 0; r < 8; ++r) s[nt].f[r] = 0.0f;
            const int col = kvb + nt * 16 + l15;
#pragma unroll
            for (int kk = 0; kk < 2; ++kk) {
                FragB16 bk;
#pragma unroll
                for (int j = 0; j < 8; ++j) {
                    const int d = kk * 32 + b_kidx(j, lh);
                    bk.u[j] = *(const unsigned*)&Kh[(size_t)col * HD + d];
                }
                s[nt].v = __builtin_amdgcn_wmma_f32_16x16x32_bf16(
                    false, aq[kk].v, false, bk.v, (short)0, s[nt].v, false, false);
            }
        }

        // ---- causal mask + scale (1/sqrt(64) = 0.125) ----
#pragma unroll
        for (int nt = 0; nt < 4; ++nt)
#pragma unroll
            for (int r = 0; r < 8; ++r) {
                const int row = q0 + wave * 16 + r + 8 * lh;
                const int col = kvb + nt * 16 + l15;
                float v = s[nt].f[r];
                if (col > row) v = -1.0e9f;
                s[nt].f[r] = v * 0.125f;
            }

        // ---- online softmax: row reductions via 16-lane xor shuffles ----
#pragma unroll
        for (int r = 0; r < 8; ++r) {
            float mx = s[0].f[r];
#pragma unroll
            for (int nt = 1; nt < 4; ++nt) mx = fmaxf(mx, s[nt].f[r]);
            mx = fmaxf(mx, __shfl_xor(mx, 1));
            mx = fmaxf(mx, __shfl_xor(mx, 2));
            mx = fmaxf(mx, __shfl_xor(mx, 4));
            mx = fmaxf(mx, __shfl_xor(mx, 8));
            const float m2 = fmaxf(m[r], mx);
            const float alpha = __expf(m[r] - m2);
            m[r] = m2;

            float rs = 0.0f;
#pragma unroll
            for (int nt = 0; nt < 4; ++nt) {
                const float p = __expf(s[nt].f[r] - m2);
                s[nt].f[r] = p;
                rs += p;
            }
            rs += __shfl_xor(rs, 1);
            rs += __shfl_xor(rs, 2);
            rs += __shfl_xor(rs, 4);
            rs += __shfl_xor(rs, 8);
            l[r] = l[r] * alpha + rs;
#pragma unroll
            for (int nt = 0; nt < 4; ++nt) o[nt].f[r] *= alpha;
        }

        // ---- stage P (C-layout) into LDS, re-read in A-layout ----
#pragma unroll
        for (int nt = 0; nt < 4; ++nt)
#pragma unroll
            for (int r = 0; r < 8; ++r)
                sP[wave][(r + 8 * lh) * 64 + nt * 16 + l15] = (__bf16)s[nt].f[r];
        asm volatile("s_wait_dscnt 0" ::: "memory");  // same-wave LDS RAW

        // ---- ctx += P @ V (64 kv positions -> two k-steps) ----
#pragma unroll
        for (int kk = 0; kk < 2; ++kk) {
            FragB16 ap;
#pragma unroll
            for (int j = 0; j < 8; ++j) {
                const int kc = kk * 32 + a_kidx(j, lh);
                ap.u[j] = *(const unsigned*)&sP[wave][l15 * 64 + kc];
            }
#pragma unroll
            for (int nt = 0; nt < 4; ++nt) {
                FragB16 bv;
#pragma unroll
                for (int j = 0; j < 8; ++j) {
                    const int sidx = kvb + kk * 32 + b_kidx(j, lh);
                    bv.u[j] = *(const unsigned*)&Vh[(size_t)(nt * 16 + l15) * SS + sidx];
                }
                o[nt].v = __builtin_amdgcn_wmma_f32_16x16x32_bf16(
                    false, ap.v, false, bv.v, (short)0, o[nt].v, false, false);
            }
        }
    }

    // ---- normalize and store ctx[b][s][h*64+d] ----
#pragma unroll
    for (int nt = 0; nt < 4; ++nt)
#pragma unroll
        for (int r = 0; r < 8; ++r) {
            const int row = q0 + wave * 16 + r + 8 * lh;
            const int col = h * HD + nt * 16 + l15;
            Ctx[((size_t)b * SS + row) * DOUT + col] = (__bf16)(o[nt].f[r] / l[r]);
        }
}

// ---------------------------------------------------------------------------
// Host-side launch
// ---------------------------------------------------------------------------
extern "C" void kernel_launch(void* const* d_in, const int* in_sizes, int n_in,
                              void* d_out, int out_size, void* d_ws, size_t ws_size,
                              hipStream_t stream) {
    const float* x    = (const float*)d_in[0];
    // d_in[1] = mask (causal triu k=1) — applied analytically in attn_kernel.
    const float* cosT = (const float*)d_in[2];
    const float* sinT = (const float*)d_in[3];
    const float* Wq   = (const float*)d_in[4];
    const float* Wk   = (const float*)d_in[5];
    const float* Wv   = (const float*)d_in[6];
    const float* Wo   = (const float*)d_in[7];

    // Workspace layout (bf16 elements). Weights stored TRANSPOSED [N][K].
    __bf16* ws = (__bf16*)d_ws;
    size_t off = 0;
    __bf16* Xbf  = ws + off; off += (size_t)BB * SS * DIN;
    __bf16* Wqt  = ws + off; off += (size_t)DIN * DOUT;
    __bf16* Wkt  = ws + off; off += (size_t)DIN * NKV * HD;
    __bf16* Wvt  = ws + off; off += (size_t)DIN * NKV * HD;
    __bf16* Wot  = ws + off; off += (size_t)DOUT * DOUT;
    __bf16* Qraw = ws + off; off += (size_t)BB * SS * DOUT;      // reused as Ctx
    __bf16* Kraw = ws + off; off += (size_t)BB * SS * NKV * HD;
    __bf16* Vraw = ws + off; off += (size_t)BB * SS * NKV * HD;
    __bf16* Qr   = ws + off; off += (size_t)BB * NH * SS * HD;
    __bf16* Kr   = ws + off; off += (size_t)BB * NKV * SS * HD;
    __bf16* Vt   = ws + off; off += (size_t)BB * NKV * HD * SS;
    __bf16* Ctx  = Qraw;  // Qraw dead after RoPE reshape

    const int M = BB * SS;  // 4096 rows

    // 1) fp32 -> bf16 (x straight; weights transposed for TDM-friendly tiles)
    {
        long n4 = (long)M * DIN / 4;
        f32_to_bf16_kernel<<<(n4 + 255) / 256, 256, 0, stream>>>(x, Xbf, n4);
    }
    f32_to_bf16_transpose_kernel<<<dim3(DOUT / 32, DIN / 32), 256, 0, stream>>>(Wq, Wqt, DIN, DOUT);
    f32_to_bf16_transpose_kernel<<<dim3((NKV * HD) / 32, DIN / 32), 256, 0, stream>>>(Wk, Wkt, DIN, NKV * HD);
    f32_to_bf16_transpose_kernel<<<dim3((NKV * HD) / 32, DIN / 32), 256, 0, stream>>>(Wv, Wvt, DIN, NKV * HD);
    f32_to_bf16_transpose_kernel<<<dim3(DOUT / 32, DOUT / 32), 256, 0, stream>>>(Wo, Wot, DOUT, DOUT);

    // 2) QKV projections (TDM-staged bf16 WMMA GEMM)
    gemm_bf16_wmma<false><<<dim3(DOUT / 64, M / 128), 256, 0, stream>>>(
        Xbf, Wqt, Qraw, M, DOUT, DIN);
    gemm_bf16_wmma<false><<<dim3((NKV * HD) / 64, M / 128), 256, 0, stream>>>(
        Xbf, Wkt, Kraw, M, NKV * HD, DIN);
    gemm_bf16_wmma<false><<<dim3((NKV * HD) / 64, M / 128), 256, 0, stream>>>(
        Xbf, Wvt, Vraw, M, NKV * HD, DIN);

    // 3) RoPE + head reshape; V transpose
    {
        long nq = (long)BB * NH * SS * HD;
        rope_reshape_kernel<<<(nq + 255) / 256, 256, 0, stream>>>(Qraw, cosT, sinT, Qr, NH);
        long nk = (long)BB * NKV * SS * HD;
        rope_reshape_kernel<<<(nk + 255) / 256, 256, 0, stream>>>(Kraw, cosT, sinT, Kr, NKV);
        v_transpose_kernel<<<(nk + 255) / 256, 256, 0, stream>>>(Vraw, Vt);
    }

    // 4) Flash-style causal attention -> Ctx (bf16)
    attn_kernel<<<dim3(SS / 64, NH, BB), 128, 0, stream>>>(Qr, Kr, Vt, Ctx);

    // 5) Output projection -> fp32 d_out
    gemm_bf16_wmma<true><<<dim3(DOUT / 64, M / 128), 256, 0, stream>>>(
        Ctx, Wot, d_out, M, DOUT, DOUT);
}
